// Net_gcn_59768764891999
// MI455X (gfx1250) — matrix operand order repaired and checked by reading, not compiled, not run on hardware
//
#include <hip/hip_runtime.h>
#include <hip/hip_bf16.h>
#include <math.h>

typedef __attribute__((ext_vector_type(16))) _Float16 v16h;
typedef __attribute__((ext_vector_type(8)))  float    v8f;

#define F_IN 512
#define F_H  16
#define F_C  10
#define KSTEPS (F_IN / 32)          // 16
#define TILES_PER_BLOCK 8           // 8 waves * 16 rows = 128 rows per block
#define BF_WORDS (KSTEPS * 32 * 8)  // 4096 dwords = 16 KB packed-f16 B frags

// ---------------------------------------------------------------- degree init
__global__ void k_init_deg(float* __restrict__ deg, int N) {
    int i = blockIdx.x * blockDim.x + threadIdx.x;
    if (i < N) deg[i] = 1.0f;                       // self-loop weight
}

// deg[col[e]] += ew[e]
__global__ void k_deg_edges(const int* __restrict__ cols,
                            const float* __restrict__ ew,
                            float* __restrict__ deg, int E) {
    int e = blockIdx.x * blockDim.x + threadIdx.x;
    if (e < E) atomicAdd(&deg[cols[e]], ew[e]);
}

// deg -> dinv in place
__global__ void k_dinv(float* __restrict__ deg, int N) {
    int i = blockIdx.x * blockDim.x + threadIdx.x;
    if (i < N) {
        float d = deg[i];
        deg[i] = (d > 0.0f) ? rsqrtf(d) : 0.0f;
    }
}

// -------------------------------------------------- h1 = x @ W1 via WMMA f16
// 8 waves/block, one 16-row tile per wave; B fragments staged once in LDS.
// Dual accumulators break the WMMA->WMMA RAW chain (no hazard v_nops).
__global__ __launch_bounds__(256) void k_gemm1_wmma(
        const float* __restrict__ x, const float* __restrict__ W1,
        float* __restrict__ h1, int N) {
    __shared__ unsigned int Bf[BF_WORDS];

    // Cooperative build of packed-f16 B fragments (shared by all waves):
    // Bf[(ks*32 + lane)*8 + v] = pack(W1[kb][ln], W1[kb+1][ln])
    // with hi = lane>>4, ln = lane&15, kb = ks*32 + 2v + hi*8 + (v&4)*2
    for (int idx = threadIdx.x; idx < BF_WORDS; idx += blockDim.x) {
        int v    = idx & 7;
        int lane = (idx >> 3) & 31;
        int ks   = idx >> 8;
        int hi = lane >> 4, ln = lane & 15;
        int kb = ks * 32 + 2 * v + hi * 8 + (v & 4) * 2;
        union { _Float16 h[2]; unsigned int u; } p;
        p.h[0] = (_Float16)W1[(size_t)kb * F_H + ln];
        p.h[1] = (_Float16)W1[(size_t)(kb + 1) * F_H + ln];
        Bf[idx] = p.u;
    }
    __syncthreads();

    const int lane = threadIdx.x & 31;
    const int wave = threadIdx.x >> 5;
    const int tile = blockIdx.x * TILES_PER_BLOCK + wave;
    const int m0   = tile * 16;
    if (m0 >= N) return;             // wave-uniform: EXEC stays full for WMMA

    const int hi = lane >> 4;        // lane half (0: lanes 0-15, 1: 16-31)
    const int ln = lane & 15;

    int am = m0 + ln;                // A-matrix row carried by this lane
    if (am >= N) am = N - 1;         // clamp tail (stores guarded below)
    const float* __restrict__ arow = x + (size_t)am * F_IN;

    v8f acc0 = {};
    v8f acc1 = {};
    #pragma unroll 4
    for (int kk = 0; kk < F_IN; kk += 32) {
        // Speculative prefetch 4 k-steps ahead (overrun is harmless/dropped).
        __builtin_prefetch(arow + kk + 128, 0, 1);      // global_prefetch_b8

        // A fragment (16-bit A 16x32 layout, ISA 7.12.2). The 8 f16-pairs of
        // this lane cover two contiguous 8-float runs -> 4x b128 loads.
        const float4* ap0 = (const float4*)(arow + kk + hi * 8);
        const float4* ap1 = (const float4*)(arow + kk + 16 + hi * 8);
        float4 f0 = ap0[0], f1 = ap0[1];
        float4 f2 = ap1[0], f3 = ap1[1];
        v16h a;
        a[0]  = (_Float16)f0.x; a[1]  = (_Float16)f0.y;
        a[2]  = (_Float16)f0.z; a[3]  = (_Float16)f0.w;
        a[4]  = (_Float16)f1.x; a[5]  = (_Float16)f1.y;
        a[6]  = (_Float16)f1.z; a[7]  = (_Float16)f1.w;
        a[8]  = (_Float16)f2.x; a[9]  = (_Float16)f2.y;
        a[10] = (_Float16)f2.z; a[11] = (_Float16)f2.w;
        a[12] = (_Float16)f3.x; a[13] = (_Float16)f3.y;
        a[14] = (_Float16)f3.z; a[15] = (_Float16)f3.w;

        // B fragment: two ds_load_b128 of the pre-packed tile
        const uint4* bp = (const uint4*)&Bf[((kk >> 5) * 32 + lane) * 8];
        union { uint4 q[2]; v16h h; } bv;
        bv.q[0] = bp[0];
        bv.q[1] = bp[1];

        if ((kk >> 5) & 1)
            acc1 = __builtin_amdgcn_wmma_f32_16x16x32_f16(
                       false, a, false, bv.h, (short)0, acc1, false, false);
        else
            acc0 = __builtin_amdgcn_wmma_f32_16x16x32_f16(
                       false, a, false, bv.h, (short)0, acc0, false, false);
    }
    v8f acc = acc0 + acc1;

    // C/D layout: VGPR r -> M = r + hi*8, N = ln
    if (m0 + 16 <= N) {              // fast path: unconditional stores
        #pragma unroll
        for (int r = 0; r < 8; ++r)
            h1[(size_t)(m0 + r + hi * 8) * F_H + ln] = acc[r];
    } else {
        #pragma unroll
        for (int r = 0; r < 8; ++r) {
            int m = m0 + r + hi * 8;
            if (m < N) h1[(size_t)m * F_H + ln] = acc[r];
        }
    }
}

// ---------------------- x1 init: b1 + self-loop term dinv[i]^2 * h1[i]
__global__ void k_init_x1(const float* __restrict__ h1,
                          const float* __restrict__ dinv,
                          const float* __restrict__ b1,
                          float* __restrict__ x1, int N) {
    int i = blockIdx.x * blockDim.x + threadIdx.x;
    if (i >= N) return;
    float s = dinv[i] * dinv[i];
    const float* hr = h1 + (size_t)i * F_H;
    float*       xr = x1 + (size_t)i * F_H;
    #pragma unroll
    for (int f = 0; f < F_H; ++f) xr[f] = b1[f] + s * hr[f];
}

// ---------------------- layer-1 edge scatter: x1[col] += norm * h1[row]
__global__ void k_scatter1(const int* __restrict__ rows,
                           const int* __restrict__ cols,
                           const float* __restrict__ ew,
                           const float* __restrict__ dinv,
                           const float* __restrict__ h1,
                           float* __restrict__ x1, int E) {
    int e = blockIdx.x * blockDim.x + threadIdx.x;
    if (e >= E) return;
    int r = rows[e], c = cols[e];
    float nrm = dinv[r] * ew[e] * dinv[c];
    const float4* hr = (const float4*)(h1 + (size_t)r * F_H);
    float* out = x1 + (size_t)c * F_H;
    #pragma unroll
    for (int q = 0; q < 4; ++q) {
        float4 v = hr[q];                       // L2-resident gather
        atomicAdd(out + 4 * q + 0, nrm * v.x);
        atomicAdd(out + 4 * q + 1, nrm * v.y);
        atomicAdd(out + 4 * q + 2, nrm * v.z);
        atomicAdd(out + 4 * q + 3, nrm * v.w);
    }
}

// ------------- h2 = relu(x1) @ W2 ; logits init = b2 + dinv^2 * h2 (self-loop)
__global__ void k_node2(const float* __restrict__ x1,
                        const float* __restrict__ W2,
                        const float* __restrict__ b2,
                        const float* __restrict__ dinv,
                        float* __restrict__ h2,
                        float* __restrict__ logits, int N) {
    int i = blockIdx.x * blockDim.x + threadIdx.x;
    if (i >= N) return;
    float r[F_H];
    const float4* xr = (const float4*)(x1 + (size_t)i * F_H);
    #pragma unroll
    for (int q = 0; q < 4; ++q) {
        float4 v = xr[q];
        r[4 * q + 0] = fmaxf(v.x, 0.0f);
        r[4 * q + 1] = fmaxf(v.y, 0.0f);
        r[4 * q + 2] = fmaxf(v.z, 0.0f);
        r[4 * q + 3] = fmaxf(v.w, 0.0f);
    }
    float sl = dinv[i] * dinv[i];
    #pragma unroll
    for (int c = 0; c < F_C; ++c) {
        float s = 0.0f;
        #pragma unroll
        for (int k = 0; k < F_H; ++k) s = fmaf(r[k], W2[k * F_C + c], s);
        h2[(size_t)i * F_C + c]     = s;
        logits[(size_t)i * F_C + c] = b2[c] + sl * s;
    }
}

// ---------------------- layer-2 edge scatter: logits[col] += norm * h2[row]
__global__ void k_scatter2(const int* __restrict__ rows,
                           const int* __restrict__ cols,
                           const float* __restrict__ ew,
                           const float* __restrict__ dinv,
                           const float* __restrict__ h2,
                           float* __restrict__ logits, int E) {
    int e = blockIdx.x * blockDim.x + threadIdx.x;
    if (e >= E) return;
    int r = rows[e], c = cols[e];
    float nrm = dinv[r] * ew[e] * dinv[c];
    const float* hr = h2 + (size_t)r * F_C;
    float* out = logits + (size_t)c * F_C;
    #pragma unroll
    for (int q = 0; q < F_C; ++q) atomicAdd(out + q, nrm * hr[q]);
}

// ---------------------- row-wise log_softmax in place on logits
__global__ void k_logsoftmax(float* __restrict__ logits, int N) {
    int i = blockIdx.x * blockDim.x + threadIdx.x;
    if (i >= N) return;
    float* row = logits + (size_t)i * F_C;
    float v[F_C];
    float m = -INFINITY;
    #pragma unroll
    for (int c = 0; c < F_C; ++c) { v[c] = row[c]; m = fmaxf(m, v[c]); }
    float s = 0.0f;
    #pragma unroll
    for (int c = 0; c < F_C; ++c) s += __expf(v[c] - m);
    float lse = m + __logf(s);
    #pragma unroll
    for (int c = 0; c < F_C; ++c) row[c] = v[c] - lse;
}

extern "C" void kernel_launch(void* const* d_in, const int* in_sizes, int n_in,
                              void* d_out, int out_size, void* d_ws, size_t ws_size,
                              hipStream_t stream) {
    const float* x  = (const float*)d_in[0];
    const int*   ei = (const int*)  d_in[1];
    const float* ew = (const float*)d_in[2];
    const float* W1 = (const float*)d_in[3];
    const float* b1 = (const float*)d_in[4];
    const float* W2 = (const float*)d_in[5];
    const float* b2 = (const float*)d_in[6];

    const int N = in_sizes[0] / F_IN;     // 100000
    const int E = in_sizes[1] / 2;        // 3200000
    const int* rows = ei;
    const int* cols = ei + E;

    float* ws   = (float*)d_ws;
    float* dinv = ws;                         // N floats (deg -> dinv in place)
    float* h1   = ws + (size_t)N;             // N*16
    float* h2   = ws + (size_t)N * 17;        // N*10

    float* out    = (float*)d_out;
    float* logits = out;                      // N*10 (output 0)
    float* x1     = out + (size_t)N * F_C;    // N*16 (output 1)

    const int TB = 256;
    const int nbN = (N + TB - 1) / TB;
    const int nbE = (E + TB - 1) / TB;
    const int tiles  = (N + 15) / 16;
    const int nbGemm = (tiles + TILES_PER_BLOCK - 1) / TILES_PER_BLOCK;

    k_init_deg  <<<nbN, TB, 0, stream>>>(dinv, N);
    k_deg_edges <<<nbE, TB, 0, stream>>>(cols, ew, dinv, E);
    k_dinv      <<<nbN, TB, 0, stream>>>(dinv, N);
    k_gemm1_wmma<<<nbGemm, 256, 0, stream>>>(x, W1, h1, N);
    k_init_x1   <<<nbN, TB, 0, stream>>>(h1, dinv, b1, x1, N);
    k_scatter1  <<<nbE, TB, 0, stream>>>(rows, cols, ew, dinv, h1, x1, E);
    k_node2     <<<nbN, TB, 0, stream>>>(x1, W2, b2, dinv, h2, logits, N);
    k_scatter2  <<<nbE, TB, 0, stream>>>(rows, cols, ew, dinv, h2, logits, E);
    k_logsoftmax<<<nbN, TB, 0, stream>>>(logits, N);
}
